// VICRegL_Loss_54889682043514
// MI455X (gfx1250) — matrix operand order repaired and checked by reading, not compiled, not run on hardware
//
#include <hip/hip_runtime.h>
#include <hip/hip_bf16.h>

// VICRegL loss for emb shape (16,1,128,32,32): Bn=16, M=1024 tokens, C=128 ch.
// Pipeline: prep (bf16 transpose-pack + norms) -> WMMA NN argmin (x2, software
// pipelined double-buffered B tiles) -> per-position vicreg partials (f32,
// Gram-Frobenius trick) -> finalize.

#define BN 16
#define MTOK 1024
#define CCH 128

typedef __attribute__((ext_vector_type(16))) __bf16 v16bf;
typedef __attribute__((ext_vector_type(8)))  float  v8f;

__device__ __forceinline__ unsigned f2bf(float v) {
    unsigned u = __float_as_uint(v);
    u += 0x7FFFu + ((u >> 16) & 1u);   // round-to-nearest-even
    return u >> 16;
}

__device__ __forceinline__ float blockReduceSum(float v, float* red) {
    #pragma unroll
    for (int off = 16; off >= 1; off >>= 1) v += __shfl_down(v, off, 32);
    __syncthreads();
    if ((threadIdx.x & 31) == 0) red[threadIdx.x >> 5] = v;
    __syncthreads();
    float s = 0.f;
    const int nw = blockDim.x >> 5;
    for (int i = 0; i < nw; ++i) s += red[i];
    return s;
}

// ---------------------------------------------------------------------------
// Kernel P: for each token, pack 128 channels as bf16 token-major + f32 norm.
// Reads are coalesced along the token dim (fastest-varying in the input).
// grid (4, 16, 2) x 256 threads; z selects emb1/emb2.
// ---------------------------------------------------------------------------
__global__ void __launch_bounds__(256) prep_kernel(
        const float* __restrict__ e1, const float* __restrict__ e2,
        unsigned short* __restrict__ t1, unsigned short* __restrict__ t2,
        float* __restrict__ n1, float* __restrict__ n2) {
    const int which = blockIdx.z;
    const float* src = which ? e2 : e1;
    unsigned short* dst = which ? t2 : t1;
    float* nrm = which ? n2 : n1;
    const int b = blockIdx.y;
    const int m = blockIdx.x * 256 + threadIdx.x;
    const float* base = src + (size_t)b * CCH * MTOK + m;
    unsigned buf[CCH / 2];
    float acc = 0.f;
    #pragma unroll
    for (int c = 0; c < CCH; c += 2) {
        float v0 = base[(size_t)c * MTOK];
        float v1 = base[(size_t)(c + 1) * MTOK];
        acc += v0 * v0 + v1 * v1;
        buf[c >> 1] = f2bf(v0) | (f2bf(v1) << 16);
    }
    nrm[b * MTOK + m] = acc;
    uint4* o = reinterpret_cast<uint4*>(dst + ((size_t)b * MTOK + m) * CCH);
    #pragma unroll
    for (int i = 0; i < CCH / 8; ++i)
        o[i] = make_uint4(buf[4 * i], buf[4 * i + 1], buf[4 * i + 2], buf[4 * i + 3]);
}

// ---------------------------------------------------------------------------
// Kernel G: idx[b,m] = argmin_n ( ||Y_n||^2 - 2 * X_m . Y_n ).
// One wave owns a 16-row stripe; K=128 handled as 4 x v_wmma_f32_16x16x32_bf16.
// A fragment (16x32 bf16) loaded once, reused across all 64 N tiles.
// B tiles are ping-pong double-buffered: all 8 b128 loads of tile t+1 issue
// before the 4 WMMAs of tile t, hiding L2 latency behind matrix ops.
// grid (16 rowgroups, 16 batches) x 128 threads (4 waves).
// ---------------------------------------------------------------------------
union CVbf { unsigned u; __bf16 h[2]; };

__device__ __forceinline__ void load_btile(const unsigned* __restrict__ Y32,
                                           size_t nrow, int half, v16bf* bf) {
    // B 32x16 layout: lane holds col N=l15; K as A: r<4 -> 8*half + 2r, r>=4 -> 16 + 8*half + 2(r-4) (+kb*32)
    // -> per lane: two runs of 4 consecutive dwords per fragment (2x b128).
    #pragma unroll
    for (int kb = 0; kb < 4; ++kb) {
        #pragma unroll
        for (int r = 0; r < 8; ++r) {
            const int kk = kb * 32 + ((r >= 4) ? 16 : 0) + 8 * half + 2 * (r & 3);
            CVbf cv; cv.u = Y32[nrow + (kk >> 1)];
            bf[kb][2 * r]     = cv.h[0];
            bf[kb][2 * r + 1] = cv.h[1];
        }
    }
}

__device__ __forceinline__ void process_tile(const v16bf* __restrict__ afrag,
                                             const v16bf* __restrict__ bfrag,
                                             float ny, int n,
                                             v8f& best, int* bidx) {
    v8f acc = {0.f, 0.f, 0.f, 0.f, 0.f, 0.f, 0.f, 0.f};
    #pragma unroll
    for (int kb = 0; kb < 4; ++kb)
        acc = __builtin_amdgcn_wmma_f32_16x16x32_bf16(
                  false, afrag[kb], false, bfrag[kb], (short)0, acc, false, false);
    #pragma unroll
    for (int r = 0; r < 8; ++r) {   // C/D: VGPR r = row r + 8*half, col = l15
        const float score = ny - 2.0f * acc[r];
        if (score < best[r]) { best[r] = score; bidx[r] = n; }
    }
}

__global__ void __launch_bounds__(128) nn_kernel(
        const unsigned short* __restrict__ XT,   // (BN, MTOK, CCH) bf16 token-major
        const unsigned short* __restrict__ YT,
        const float* __restrict__ normY,         // (BN, MTOK)
        int* __restrict__ idxOut) {              // (BN, MTOK)
    const int b    = blockIdx.y;
    const int wave = threadIdx.x >> 5;
    const int lane = threadIdx.x & 31;
    const int half = lane >> 4;
    const int l15  = lane & 15;
    const int m0   = blockIdx.x * 64 + wave * 16;

    const unsigned* X32 = reinterpret_cast<const unsigned*>(XT);
    const unsigned* Y32 = reinterpret_cast<const unsigned*>(YT);

    // A fragments: ISA 16-bit A 16x32 layout. lane holds row M=l15 (both halves);
    // VGPR r<4 -> K = 8*half + 2r ; r>=4 -> K = 16 + 8*half + 2(r-4).
    v16bf afrag[4];
    {
        const size_t rowbase = ((size_t)b * MTOK + (m0 + l15)) * (CCH / 2);
        #pragma unroll
        for (int kb = 0; kb < 4; ++kb) {
            #pragma unroll
            for (int r = 0; r < 8; ++r) {
                const int kk = kb * 32 + ((r >= 4) ? 16 : 0) + 8 * half + 2 * (r & 3);
                CVbf cv; cv.u = X32[rowbase + (kk >> 1)];
                afrag[kb][2 * r]     = cv.h[0];
                afrag[kb][2 * r + 1] = cv.h[1];
            }
        }
    }

    v8f best; int bidx[8];
    #pragma unroll
    for (int r = 0; r < 8; ++r) { best[r] = 3.0e38f; bidx[r] = 0; }

    const size_t nrow0   = ((size_t)b * MTOK + l15) * (CCH / 2);
    const size_t rowstep = (size_t)16 * (CCH / 2);
    const float* nyBase  = normY + b * MTOK + l15;

    v16bf b0[4], b1[4];
    load_btile(Y32, nrow0, half, b0);            // prologue: tile 0
    float ny0 = nyBase[0], ny1;

    for (int nt = 0; nt < 64; nt += 2) {
        // warm L0/L2 a few tiles ahead (global_prefetch_b8)
        if (nt + 4 < 64)
            __builtin_prefetch((const void*)(Y32 + nrow0 + (nt + 4) * rowstep), 0, 3);

        // issue loads for tile nt+1, then compute tile nt
        load_btile(Y32, nrow0 + (size_t)(nt + 1) * rowstep, half, b1);
        ny1 = nyBase[(nt + 1) * 16];
        process_tile(afrag, b0, ny0, nt * 16 + l15, best, bidx);

        // issue loads for tile nt+2, then compute tile nt+1
        if (nt + 2 < 64) {
            load_btile(Y32, nrow0 + (size_t)(nt + 2) * rowstep, half, b0);
            ny0 = nyBase[(nt + 2) * 16];
        }
        process_tile(afrag, b1, ny1, (nt + 1) * 16 + l15, best, bidx);
    }

    // reduce across the 16 lanes of each half (masks < 16 stay in-half)
    #pragma unroll
    for (int r = 0; r < 8; ++r) {
        float bv = best[r]; int bi = bidx[r];
        #pragma unroll
        for (int s = 8; s >= 1; s >>= 1) {
            const float ov = __shfl_xor(bv, s, 32);
            const int   oi = __shfl_xor(bi, s, 32);
            if (ov < bv || (ov == bv && oi < bi)) { bv = ov; bi = oi; }
        }
        if (l15 == 0) idxOut[b * MTOK + m0 + r + 8 * half] = bi;
    }
}

// ---------------------------------------------------------------------------
// Kernel V: per (position, direction) vicreg partial sums in exact f32.
// Uses ||A^T A||_F == ||A A^T||_F : only a 16x16 Gram per position needed.
// grid (1024, 2) x 128 threads (thread = channel).
// ---------------------------------------------------------------------------
__global__ void __launch_bounds__(128) vicreg_kernel(
        const float* __restrict__ e1, const float* __restrict__ e2,
        const int* __restrict__ idx1, const int* __restrict__ idx2,
        float* __restrict__ partials) {
    const int m    = blockIdx.x;
    const int call = blockIdx.y;
    const float* X = call ? e2 : e1;
    const float* Y = call ? e1 : e2;
    const int* idx = call ? idx2 : idx1;
    const int c = threadIdx.x;

    __shared__ int   sIdx[BN];
    __shared__ float ldsX[BN][CCH + 4];
    __shared__ float ldsY[BN][CCH + 4];
    __shared__ float red[8];

    if (c < BN) sIdx[c] = idx[c * MTOK + m];
    __syncthreads();

    float xv[BN], yv[BN];
    float repr = 0.f, mx = 0.f, my = 0.f;
    #pragma unroll
    for (int b = 0; b < BN; ++b) {
        xv[b] = X[((size_t)b * CCH + c) * MTOK + m];
        yv[b] = Y[((size_t)b * CCH + c) * MTOK + sIdx[b]];
        const float d = xv[b] - yv[b];
        repr += d * d;
        mx += xv[b]; my += yv[b];
    }
    mx *= (1.f / 16.f); my *= (1.f / 16.f);
    float sx = 0.f, sy = 0.f;
    #pragma unroll
    for (int b = 0; b < BN; ++b) {
        xv[b] -= mx; yv[b] -= my;
        sx += xv[b] * xv[b]; sy += yv[b] * yv[b];
        ldsX[b][c] = xv[b];  ldsY[b][c] = yv[b];
    }
    const float stdx = sqrtf(sx * (1.f / 15.f) + 1e-4f);
    const float stdy = sqrtf(sy * (1.f / 15.f) + 1e-4f);
    const float relus = fmaxf(0.f, 1.f - stdx) + fmaxf(0.f, 1.f - stdy);
    const float ss2 = sx * sx + sy * sy;     // per-channel diag^2 terms (x+y)
    __syncthreads();

    // 16x16 Gram Frobenius norms: 136 unique (i<=j) pairs over 128 threads
    float kacc = 0.f;
    for (int p = c; p < 136; p += 128) {
        int i = 0, q = p;
        while (q >= BN - i) { q -= BN - i; ++i; }
        const int j = i + q;
        float dx = 0.f, dy = 0.f;
        for (int cc = 0; cc < CCH; ++cc) {
            dx += ldsX[i][cc] * ldsX[j][cc];
            dy += ldsY[i][cc] * ldsY[j][cc];
        }
        const float w = (i == j) ? 1.f : 2.f;
        kacc += w * (dx * dx + dy * dy);
    }

    const float r0 = blockReduceSum(repr, red);
    const float r1 = blockReduceSum(relus, red);
    const float r2 = blockReduceSum(kacc - ss2, red);
    if (threadIdx.x == 0) {
        float* p = partials + ((size_t)call * MTOK + m) * 3;
        p[0] = r0;
        p[1] = r1;
        p[2] = r2 * (1.f / (225.f * 256.f));   // /(n-1)^2, /C, /2
    }
}

// ---------------------------------------------------------------------------
// Kernel R: deterministic final reduction (a == b, so final loss == a).
// ---------------------------------------------------------------------------
__global__ void __launch_bounds__(256) finalize_kernel(
        const float* __restrict__ partials, float* __restrict__ out) {
    __shared__ float red[8];
    float s0 = 0.f, s1 = 0.f, s2 = 0.f;
    for (int i = threadIdx.x; i < 2 * MTOK; i += blockDim.x) {
        s0 += partials[i * 3 + 0];
        s1 += partials[i * 3 + 1];
        s2 += partials[i * 3 + 2];
    }
    s0 = blockReduceSum(s0, red);
    s1 = blockReduceSum(s1, red);
    s2 = blockReduceSum(s2, red);
    if (threadIdx.x == 0) {
        out[0] = 25.f * s0 / 4194304.f;  // INV: 25 * sums / (2 * 16*1024*128)
        out[1] = 25.f * s1 / 524288.f;   // VAR: 25 * sums / (4 * 1024*128)
        out[2] = s2 / 2048.f;            // COV: mean over 2*1024 (pos, dir)
    }
}

extern "C" void kernel_launch(void* const* d_in, const int* in_sizes, int n_in,
                              void* d_out, int out_size, void* d_ws, size_t ws_size,
                              hipStream_t stream) {
    (void)in_sizes; (void)n_in; (void)out_size; (void)ws_size;
    const float* e1 = (const float*)d_in[0];
    const float* e2 = (const float*)d_in[1];
    float* out = (float*)d_out;

    char* ws = (char*)d_ws;
    size_t off = 0;
    auto alloc = [&](size_t bytes) -> void* {
        void* p = ws + off;
        off += (bytes + 255) & ~(size_t)255;
        return p;
    };
    unsigned short* t1 = (unsigned short*)alloc((size_t)BN * MTOK * CCH * 2);
    unsigned short* t2 = (unsigned short*)alloc((size_t)BN * MTOK * CCH * 2);
    float* n1 = (float*)alloc((size_t)BN * MTOK * 4);
    float* n2 = (float*)alloc((size_t)BN * MTOK * 4);
    int*   i1 = (int*)alloc((size_t)BN * MTOK * 4);
    int*   i2 = (int*)alloc((size_t)BN * MTOK * 4);
    float* parts = (float*)alloc((size_t)2 * MTOK * 3 * 4);

    prep_kernel<<<dim3(4, 16, 2), 256, 0, stream>>>(e1, e2, t1, t2, n1, n2);
    nn_kernel<<<dim3(16, 16), 128, 0, stream>>>(t1, t2, n2, i1);  // NN of m1 in m2
    nn_kernel<<<dim3(16, 16), 128, 0, stream>>>(t2, t1, n1, i2);  // NN of m2 in m1
    vicreg_kernel<<<dim3(MTOK, 2), 128, 0, stream>>>(e1, e2, i1, i2, parts);
    finalize_kernel<<<1, 256, 0, stream>>>(parts, out);
}
